// GCNModel_58394375357080
// MI455X (gfx1250) — matrix operand order repaired and checked by reading, not compiled, not run on hardware
//
#include <hip/hip_runtime.h>

// ---------------------------------------------------------------------------
// Types for CDNA5 WMMA (wave32): 16x16x32 bf16 -> f32
// ---------------------------------------------------------------------------
typedef __attribute__((ext_vector_type(8)))  unsigned short u16x8;
typedef __attribute__((ext_vector_type(16))) unsigned short u16x16;
typedef __attribute__((ext_vector_type(16))) __bf16         v16bf;
typedef __attribute__((ext_vector_type(8)))  float          v8f;

union BFPun { u16x16 u; v16bf b; };

__device__ __forceinline__ unsigned short f32_to_bf16_rne(float f) {
    unsigned int u = __float_as_uint(f);
    u += 0x7FFFu + ((u >> 16) & 1u);           // round-to-nearest-even
    return (unsigned short)(u >> 16);
}

// ---------------------------------------------------------------------------
// Utility kernels
// ---------------------------------------------------------------------------
__global__ void k_fill(float* __restrict__ p, float v, long long n) {
    long long i = (long long)blockIdx.x * blockDim.x + threadIdx.x;
    if (i < n) p[i] = v;
}

__global__ void k_deg(const int* __restrict__ dst, float* __restrict__ degf, int E) {
    int e = blockIdx.x * blockDim.x + threadIdx.x;
    if (e < E) atomicAdd(&degf[dst[e]], 1.0f);
}

__global__ void k_rsqrt(const float* __restrict__ degf, float* __restrict__ dinv, int n) {
    int i = blockIdx.x * blockDim.x + threadIdx.x;
    if (i < n) dinv[i] = rsqrtf(fmaxf(degf[i], 1.0f));
}

__global__ void k_f32_to_bf16(const float* __restrict__ in, unsigned short* __restrict__ out,
                              long long n) {
    long long i = (long long)blockIdx.x * blockDim.x + threadIdx.x;
    if (i < n) out[i] = f32_to_bf16_rne(in[i]);
}

// Wt[n*K + k] = bf16(W[k*N + n])  : column-major bf16 weights for WMMA B frags
__global__ void k_w_transpose_bf16(const float* __restrict__ W, unsigned short* __restrict__ Wt,
                                   int K, int N) {
    int i = blockIdx.x * blockDim.x + threadIdx.x;
    if (i >= K * N) return;
    int k = i / N, n = i - k * N;              // coalesced read over n
    Wt[(size_t)n * K + k] = f32_to_bf16_rne(W[i]);
}

// ---------------------------------------------------------------------------
// WMMA GEMM: C[M,N] (f32, row-major) = A[M,K] (bf16, row-major) x Bt[N,K]^T
// Each wave computes a 16x64 strip (4 N-tiles): the A fragment is loaded ONCE
// per 32-wide K step and fed to 4 v_wmma_f32_16x16x32_bf16 against 4 B frags.
// This cuts A-matrix L2 traffic 4x vs one-tile-per-wave.
// Per-lane fragment layout uses contiguous K (k = 16*half + s), same for A and B:
//   A: lane<16 -> row M=l16, K 0..15;         lane>=16 -> K 16..31
//   B: lane<16 -> col N=l16, K 0..15;         lane>=16 -> K 16..31
//   C: lane<16 -> col l16 rows 0..7;          lane>=16 -> rows 8..15
// Requires: M % 16 == 0, N % 64 == 0, K % 32 == 0 (true for all 3 GEMMs).
// ---------------------------------------------------------------------------
__global__ void k_gemm_bf16_wmma(const unsigned short* __restrict__ A,
                                 const unsigned short* __restrict__ Bt,
                                 float* __restrict__ C,
                                 int M, int N, int K) {
    const int ngrp = N >> 6;                    // groups of 4 n-tiles (64 cols)
    const int grps = (M >> 4) * ngrp;
    const int wid = (int)(((long long)blockIdx.x * blockDim.x + threadIdx.x) >> 5);
    if (wid >= grps) return;                    // wave-uniform exit (EXEC stays all-1)
    const int m0 = (wid / ngrp) << 4;
    const int n0 = (wid % ngrp) << 6;
    const int lane = threadIdx.x & 31;
    const int half = lane >> 4;                 // 0: lanes 0-15, 1: lanes 16-31
    const int l16  = lane & 15;

    const unsigned short* arow = A  + (size_t)(m0 + l16) * K + (half << 4);
    const unsigned short* brow = Bt + (size_t)(n0 + l16) * K + (half << 4);
    const size_t bstep = (size_t)16 * K;        // 16 columns ahead in col-major Bt

    v8f acc0 = {}, acc1 = {}, acc2 = {}, acc3 = {};
    for (int k = 0; k < K; k += 32) {
        __builtin_prefetch(arow + k + 64, 0, 0);          // global_prefetch_b8
        u16x8  a0 = *(const u16x8*)(arow + k);            // K blk: +0..7   (or 16..23)
        u16x8  a1 = *(const u16x8*)(arow + k + 8);        // K blk: +8..15  (or 24..31)
        u16x16 bv0 = *(const u16x16*)(brow + k);
        u16x16 bv1 = *(const u16x16*)(brow + bstep + k);
        u16x16 bv2 = *(const u16x16*)(brow + 2 * bstep + k);
        u16x16 bv3 = *(const u16x16*)(brow + 3 * bstep + k);
        u16x16 av;
#pragma unroll
        for (int i = 0; i < 8; ++i) { av[i] = a0[i]; av[8 + i] = a1[i]; }
        BFPun ua, ub0, ub1, ub2, ub3;
        ua.u = av; ub0.u = bv0; ub1.u = bv1; ub2.u = bv2; ub3.u = bv3;
        acc0 = __builtin_amdgcn_wmma_f32_16x16x32_bf16(
                   false, ua.b, false, ub0.b, (short)0, acc0, false, false);
        acc1 = __builtin_amdgcn_wmma_f32_16x16x32_bf16(
                   false, ua.b, false, ub1.b, (short)0, acc1, false, false);
        acc2 = __builtin_amdgcn_wmma_f32_16x16x32_bf16(
                   false, ua.b, false, ub2.b, (short)0, acc2, false, false);
        acc3 = __builtin_amdgcn_wmma_f32_16x16x32_bf16(
                   false, ua.b, false, ub3.b, (short)0, acc3, false, false);
    }

    float* cp = C + (size_t)(m0 + (half << 3)) * N + n0 + l16;
#pragma unroll
    for (int v = 0; v < 8; ++v) {               // 8 contiguous 64B segs per row-pair
        float* r = cp + (size_t)v * N;
        r[0]  = acc0[v];
        r[16] = acc1[v];
        r[32] = acc2[v];
        r[48] = acc3[v];
    }
}

// ---------------------------------------------------------------------------
// Edge aggregation: agg[dst] += h[src] * dinv[src]*dinv[dst]
// F/4 threads per edge, float4 gather, f32 atomics (agg & h are L2-resident).
// ---------------------------------------------------------------------------
__global__ void k_edge_scatter(const float* __restrict__ h, const int* __restrict__ src,
                               const int* __restrict__ dst, const float* __restrict__ dinv,
                               float* __restrict__ agg, int E, int F) {
    long long t = (long long)blockIdx.x * blockDim.x + threadIdx.x;
    const int cpe = F >> 2;
    long long e = t / cpe;
    if (e >= E) return;
    int j = (int)(t - e * cpe);
    int s = src[e], d = dst[e];
    float nrm = dinv[s] * dinv[d];
    const float4 hv = ((const float4*)(h + (size_t)s * F))[j];
    float* o = agg + (size_t)d * F + (j << 2);
    atomicAdd(o + 0, hv.x * nrm);
    atomicAdd(o + 1, hv.y * nrm);
    atomicAdd(o + 2, hv.z * nrm);
    atomicAdd(o + 3, hv.w * nrm);
}

// Layer-1 epilogue: add self-loop (h/deg), bias, ReLU; emit bf16 for next GEMM.
__global__ void k_bias_relu_to_bf16(const float* __restrict__ agg, const float* __restrict__ h,
                                    const float* __restrict__ dinv, const float* __restrict__ b,
                                    unsigned short* __restrict__ obf, int Nn, int F) {
    long long i = (long long)blockIdx.x * blockDim.x + threadIdx.x;
    if (i >= (long long)Nn * F) return;
    int n = (int)(i / F), f = (int)(i - (long long)n * F);
    float di = dinv[n];
    float v = agg[i] + h[i] * di * di + b[f];
    obf[i] = f32_to_bf16_rne(fmaxf(v, 0.0f));
}

// Layer-2 epilogue fused with mean-pool accumulation (sums; counts separate).
__global__ void k_bias_relu_pool(const float* __restrict__ agg, const float* __restrict__ h,
                                 const float* __restrict__ dinv, const float* __restrict__ b,
                                 const int* __restrict__ batch, float* __restrict__ pooled,
                                 int Nn, int F) {
    long long i = (long long)blockIdx.x * blockDim.x + threadIdx.x;
    if (i >= (long long)Nn * F) return;
    int n = (int)(i / F), f = (int)(i - (long long)n * F);
    float di = dinv[n];
    float v = fmaxf(agg[i] + h[i] * di * di + b[f], 0.0f);
    atomicAdd(&pooled[(size_t)batch[n] * F + f], v);
}

__global__ void k_count(const int* __restrict__ batch, float* __restrict__ cnt, int Nn) {
    int i = blockIdx.x * blockDim.x + threadIdx.x;
    if (i < Nn) atomicAdd(&cnt[batch[i]], 1.0f);
}

__global__ void k_pool_finish(const float* __restrict__ pooled, const float* __restrict__ cnt,
                              unsigned short* __restrict__ pbf, int G, int F) {
    int i = blockIdx.x * blockDim.x + threadIdx.x;
    if (i >= G * F) return;
    int g = i / F;
    pbf[i] = f32_to_bf16_rne(pooled[i] / fmaxf(cnt[g], 1.0f));
}

__global__ void k_out_bias_relu(const float* __restrict__ fc, const float* __restrict__ fcb,
                                float* __restrict__ out, int n, int N) {
    int i = blockIdx.x * blockDim.x + threadIdx.x;
    if (i < n) out[i] = fmaxf(fc[i] + fcb[i % N], 0.0f);
}

// ---------------------------------------------------------------------------
// Host launcher
// ---------------------------------------------------------------------------
extern "C" void kernel_launch(void* const* d_in, const int* in_sizes, int n_in,
                              void* d_out, int out_size, void* d_ws, size_t ws_size,
                              hipStream_t stream) {
    const float* x    = (const float*)d_in[0];
    const int*   ei   = (const int*)d_in[1];
    const int*   batch= (const int*)d_in[2];
    const float* W1   = (const float*)d_in[3];
    const float* b1   = (const float*)d_in[4];
    const float* W2   = (const float*)d_in[5];
    const float* b2   = (const float*)d_in[6];
    const float* fcW  = (const float*)d_in[7];
    const float* fcb  = (const float*)d_in[8];

    const int IN_DIM = 256, H1 = 128, H2 = 256, OUT = 512, G = 64;
    const int Nn = in_sizes[0] / IN_DIM;       // 100000 (divisible by 16)
    const int E  = in_sizes[1] / 2;            // 1600000
    const int* src = ei;
    const int* dst = ei + E;

    char* ws = (char*)d_ws;
    size_t off = 0;
    auto carve = [&](size_t bytes) -> char* {
        char* p = ws + off;
        off = (off + bytes + 255) & ~(size_t)255;
        return p;
    };

    unsigned short* xbf  = (unsigned short*)carve((size_t)Nn * IN_DIM * 2); // also h1act later
    unsigned short* w1t  = (unsigned short*)carve((size_t)H1 * IN_DIM * 2);
    unsigned short* w2t  = (unsigned short*)carve((size_t)H2 * H1 * 2);
    unsigned short* fct  = (unsigned short*)carve((size_t)OUT * H2 * 2);
    float* degf   = (float*)carve((size_t)Nn * 4);
    float* dinv   = (float*)carve((size_t)Nn * 4);
    float* RA     = (float*)carve((size_t)Nn * H2 * 4);   // h1 (layer1) then h2 (layer2)
    float* RB     = (float*)carve((size_t)Nn * H2 * 4);   // agg1 then agg2
    float* pooled = (float*)carve((size_t)G * H2 * 4);
    float* cnt    = (float*)carve((size_t)G * 4);
    unsigned short* pbf = (unsigned short*)carve((size_t)G * H2 * 2);
    float* fco    = (float*)carve((size_t)G * OUT * 4);

    auto nb = [](long long n, int b) { return (unsigned)((n + b - 1) / b); };
    const int B = 256;

    // Degrees (self-loop folded in as init=1.0) and sym-norm factors
    k_fill <<<nb(Nn, B), B, 0, stream>>>(degf, 1.0f, (long long)Nn);
    k_deg  <<<nb(E, B), B, 0, stream>>>(dst, degf, E);
    k_rsqrt<<<nb(Nn, B), B, 0, stream>>>(degf, dinv, Nn);

    // bf16 conversions
    k_f32_to_bf16<<<nb((long long)Nn * IN_DIM, B), B, 0, stream>>>(x, xbf, (long long)Nn * IN_DIM);
    k_w_transpose_bf16<<<nb(IN_DIM * H1, B), B, 0, stream>>>(W1, w1t, IN_DIM, H1);
    k_w_transpose_bf16<<<nb(H1 * H2, B), B, 0, stream>>>(W2, w2t, H1, H2);
    k_w_transpose_bf16<<<nb(H2 * OUT, B), B, 0, stream>>>(fcW, fct, H2, OUT);

    // ---- Layer 1: h1 = x@W1 ; aggregate ; relu -> bf16 (into xbf region) ----
    long long t1 = (long long)(Nn / 16) * (H1 / 64) * 32;
    k_gemm_bf16_wmma<<<nb(t1, B), B, 0, stream>>>(xbf, w1t, RA, Nn, H1, IN_DIM);
    k_fill<<<nb((long long)Nn * H1, B), B, 0, stream>>>(RB, 0.0f, (long long)Nn * H1);
    k_edge_scatter<<<nb((long long)E * (H1 / 4), B), B, 0, stream>>>(RA, src, dst, dinv, RB, E, H1);
    k_bias_relu_to_bf16<<<nb((long long)Nn * H1, B), B, 0, stream>>>(RB, RA, dinv, b1, xbf, Nn, H1);

    // ---- Layer 2: h2 = h1act@W2 ; aggregate ; relu fused with pooling ----
    long long t2 = (long long)(Nn / 16) * (H2 / 64) * 32;
    k_gemm_bf16_wmma<<<nb(t2, B), B, 0, stream>>>(xbf, w2t, RA, Nn, H2, H1);
    k_fill<<<nb((long long)Nn * H2, B), B, 0, stream>>>(RB, 0.0f, (long long)Nn * H2);
    k_edge_scatter<<<nb((long long)E * (H2 / 4), B), B, 0, stream>>>(RA, src, dst, dinv, RB, E, H2);

    k_fill<<<nb(G * H2, B), B, 0, stream>>>(pooled, 0.0f, (long long)G * H2);
    k_fill<<<1, 64, 0, stream>>>(cnt, 0.0f, (long long)G);
    k_count<<<nb(Nn, B), B, 0, stream>>>(batch, cnt, Nn);
    k_bias_relu_pool<<<nb((long long)Nn * H2, B), B, 0, stream>>>(RB, RA, dinv, b2, batch, pooled, Nn, H2);
    k_pool_finish<<<nb(G * H2, B), B, 0, stream>>>(pooled, cnt, pbf, G, H2);

    // ---- FC: out = relu(pooled@fcW + fcb) ----
    long long t3 = (long long)(G / 16) * (OUT / 64) * 32;
    k_gemm_bf16_wmma<<<nb(t3, B), B, 0, stream>>>(pbf, fct, fco, G, OUT, H2);
    k_out_bias_relu<<<nb((long long)G * OUT, B), B, 0, stream>>>(fco, fcb, (float*)d_out, G * OUT, OUT);
}
